// SLMBlock_46660524703981
// MI455X (gfx1250) — matrix operand order repaired and checked by reading, not compiled, not run on hardware
//
#include <hip/hip_runtime.h>
#include <hip/hip_bf16.h>
#include <cstdint>
#include <cstddef>

// ---------------- CDNA5 WMMA types/helpers ----------------
typedef __attribute__((ext_vector_type(16))) __bf16 v16bf;
typedef __attribute__((ext_vector_type(8)))  float  v8f;
typedef __attribute__((ext_vector_type(4)))  unsigned u32x4;
typedef __attribute__((ext_vector_type(8)))  int      i32x8;
typedef __attribute__((ext_vector_type(4)))  int      i32x4;

#define NTOK   4096
#define HDIM   1024
#define SEQ    1024
#define BATCH  4
#define NHEADS 16
#define NKVH   4
#define HEADD  64
#define NEXP   8
#define FDIM   2048

// Tensor Data Mover availability (this toolchain: clang-23 6-arg form)
#if defined(__has_builtin)
# if __has_builtin(__builtin_amdgcn_tensor_load_to_lds) && __has_builtin(__builtin_amdgcn_s_wait_tensorcnt)
#  define USE_TDM 1
# endif
#endif
#ifndef USE_TDM
# define USE_TDM 0
#endif

union FragU { v16bf v; unsigned u[8]; };

__device__ inline v8f zero8(){
  v8f z;
  #pragma unroll
  for(int i=0;i<8;++i) z[i]=0.0f;
  return z;
}

__device__ inline unsigned short f2bf(float x){
  unsigned u = __float_as_uint(x);
  unsigned r = u + 0x7FFFu + ((u >> 16) & 1u);   // round-to-nearest-even
  return (unsigned short)(r >> 16);
}
__device__ inline unsigned packbf(float lo, float hi){
  return (unsigned)f2bf(lo) | ((unsigned)f2bf(hi) << 16);
}

// A-fragment (16x32 bf16): lanes 0-15 hold K=kb0+0..15, lanes 16-31 K=kb0+16..31.
// row = packed (k,k+1) pairs.
__device__ inline v16bf frag_a(const unsigned* row, int kb0, int lane){
  FragU f; int half = (lane >> 4) & 1; int b0 = (kb0 >> 1) + 8*half;
  #pragma unroll
  for(int j=0;j<8;++j) f.u[j]   = row[b0+j];
  return f.v;
}
// B-fragment (32x16 bf16): lanes 0-15 K=kb0..+15, lanes 16-31 K=kb0+16..+31, N=lane%16.
__device__ inline v16bf frag_b(const unsigned* col, int kb0, int lane){
  FragU f; int half = (lane >> 4) & 1; int b0 = (kb0 >> 1) + 8*half;
  #pragma unroll
  for(int j=0;j<8;++j) f.u[j] = col[b0+j];
  return f.v;
}
__device__ inline v8f wmma_bf16(v16bf a, v16bf b, v8f c){
  return __builtin_amdgcn_wmma_f32_16x16x32_bf16(false, a, false, b, (short)0, c, false, false);
}

#if USE_TDM
typedef __attribute__((address_space(3))) const char lds_char;
__device__ inline unsigned lds_off(const void* p){
  return (unsigned)(unsigned long long)(lds_char*)p;
}
// 2-D TDM tile load: 64 rows x 16 dwords (64B) into LDS, row stride in dword units.
// D# per CDNA5 ISA: group0 {count,lds,global,type=2},
// group1 {data_size=4B, dims, tile 16x64, stride0}, groups 2/3 zero (2-D tensor).
__device__ inline void tdm_load_tile64(unsigned ldsByteAddr, const unsigned* gsrc,
                                       unsigned strideElems, unsigned totalRows){
  unsigned long long ga = (unsigned long long)(const void*)gsrc;
  u32x4 g0;
  g0.x = 1u;                                                   // count=1
  g0.y = ldsByteAddr;                                          // lds_addr
  g0.z = (unsigned)ga;                                         // global_addr lo
  g0.w = ((unsigned)(ga >> 32) & 0x1FFFFFFu) | (2u << 30);     // global hi | type=2
  i32x8 g1;
  g1[0] = (int)(2u << 16);                                     // data_size = 4B
  g1[1] = (int)((strideElems & 0xFFFFu) << 16);                // tensor_dim0 lo16
  g1[2] = (int)((strideElems >> 16) | ((totalRows & 0xFFFFu) << 16));
  g1[3] = (int)((totalRows >> 16) | (16u << 16));              // tile_dim0 = 16 dwords
  g1[4] = 64;                                                  // tile_dim1 = 64 rows
  g1[5] = (int)strideElems;                                    // tensor_dim0_stride lo
  g1[6] = 0;
  g1[7] = 0;
  i32x4 z4 = {0,0,0,0};
  i32x8 z8; 
  #pragma unroll
  for(int i=0;i<8;++i) z8[i]=0;
  // clang-23 / therock-10.0 6-arg form: (g0, g1, g2, g3, extra, cpol)
  __builtin_amdgcn_tensor_load_to_lds(g0, g1, z4, z4, z8, 0);
}
#endif

// ---------------- pack-transpose converts into WMMA B-fragment layout ----------------
// W (K x N, f32) -> Wp[n*(K/2)+kp] = pack(W[2kp][n], W[2kp+1][n]); batched via blockIdx.y
__global__ void k_packT_f32(const float* __restrict__ W, unsigned* __restrict__ Wp,
                            int K, int N){
  int Kp = K >> 1;
  const float* Wb = W  + (size_t)blockIdx.y * K  * N;
  unsigned*    Pb = Wp + (size_t)blockIdx.y * Kp * N;
  int idx = blockIdx.x*blockDim.x + threadIdx.x;
  if(idx >= Kp*N) return;
  int n = idx % N, kp = idx / N;                 // coalesced reads over n
  float lo = Wb[(size_t)(2*kp    )*N + n];
  float hi = Wb[(size_t)(2*kp + 1)*N + n];
  Pb[(size_t)n*Kp + kp] = packbf(lo, hi);
}
// same, bf16 source (used to pre-pack V for attention)
__global__ void k_packT_bf16(const unsigned short* __restrict__ X, unsigned* __restrict__ Xp,
                             int K, int N){
  int Kp = K >> 1;
  const unsigned short* Xb = X  + (size_t)blockIdx.y * K  * N;
  unsigned*             Pb = Xp + (size_t)blockIdx.y * Kp * N;
  int idx = blockIdx.x*blockDim.x + threadIdx.x;
  if(idx >= Kp*N) return;
  int n = idx % N, kp = idx / N;
  unsigned lo = Xb[(size_t)(2*kp    )*N + n];
  unsigned hi = Xb[(size_t)(2*kp + 1)*N + n];
  Pb[(size_t)n*Kp + kp] = lo | (hi << 16);
}

// ---------------- RMSNorm: fp32 in, bf16 (+ optional fp32) out ----------------
__global__ void k_rmsnorm(const float* __restrict__ x, const float* __restrict__ w,
                          unsigned short* __restrict__ obf, float* __restrict__ of, int Hd){
  __shared__ float red[256];
  __shared__ float sInv;
  int row = blockIdx.x;
  const float* xr = x + (size_t)row*Hd;
  float ss = 0.0f;
  for(int i=threadIdx.x;i<Hd;i+=256){ float v = xr[i]; ss += v*v; }
  red[threadIdx.x] = ss; __syncthreads();
  for(int off=128; off>0; off>>=1){
    if((int)threadIdx.x < off) red[threadIdx.x] += red[threadIdx.x+off];
    __syncthreads();
  }
  if(threadIdx.x==0) sInv = rsqrtf(red[0]/(float)Hd + 1e-6f);
  __syncthreads();
  float inv = sInv;
  for(int i=threadIdx.x;i<Hd;i+=256)
    obf[(size_t)row*Hd + i] = f2bf(xr[i]*inv*w[i]);
  if(of){
    for(int i=threadIdx.x;i<Hd;i+=256)
      of[(size_t)row*Hd + i] = xr[i]*inv*w[i];
  }
}

// ---------------- bf16 WMMA GEMM: C(f32,MxN) = A(bf16,MxK) @ Bp(packed KxN) [+R] -------
__global__ __launch_bounds__(128)
void k_gemm(const unsigned short* __restrict__ A, const unsigned* __restrict__ Bp,
            float* __restrict__ C, const float* __restrict__ R, int M, int N, int K)
{
  __shared__ unsigned aL[2][1024];
  __shared__ unsigned bL[2][1024];
  const int tid = threadIdx.x, lane = tid & 31, wv = tid >> 5;
  const int ln = lane & 15, half = lane >> 4;
  const int rowBase = blockIdx.y*64, colBase = blockIdx.x*64;
  const int Kp = K >> 1;
  const unsigned* Au = (const unsigned*)A;
  v8f acc[4];
  #pragma unroll
  for(int i=0;i<4;++i) acc[i] = zero8();
  const int ktn = K >> 5;
  int buf = 0;
#if USE_TDM
  if(wv==0){
    tdm_load_tile64(lds_off(&aL[0][0]), Au + (size_t)rowBase*Kp, (unsigned)Kp, (unsigned)M);
    tdm_load_tile64(lds_off(&bL[0][0]), Bp + (size_t)colBase*Kp, (unsigned)Kp, (unsigned)N);
  }
#endif
  for(int kt=0; kt<ktn; ++kt){
#if USE_TDM
    if(wv==0) __builtin_amdgcn_s_wait_tensorcnt(0);
#else
    for(int i=tid;i<1024;i+=128){
      int r = i>>4, c = i&15;
      aL[0][i] = Au[(size_t)(rowBase+r)*Kp + kt*16 + c];
    }
    for(int i=tid;i<1024;i+=128){
      int n = i>>4, c = i&15;
      bL[0][i] = Bp[(size_t)(colBase+n)*Kp + kt*16 + c];
    }
#endif
    __syncthreads();
#if USE_TDM
    if(wv==0 && kt+1 < ktn){
      tdm_load_tile64(lds_off(&aL[buf^1][0]), Au + (size_t)rowBase*Kp + (kt+1)*16, (unsigned)Kp, (unsigned)M);
      tdm_load_tile64(lds_off(&bL[buf^1][0]), Bp + (size_t)colBase*Kp + (kt+1)*16, (unsigned)Kp, (unsigned)N);
    }
    const unsigned* ab = aL[buf]; const unsigned* bb = bL[buf];
#else
    const unsigned* ab = aL[0];  const unsigned* bb = bL[0];
#endif
    v16bf a = frag_a(&ab[(wv*16+ln)*16], 0, lane);
    #pragma unroll
    for(int ct=0;ct<4;++ct)
      acc[ct] = wmma_bf16(a, frag_b(&bb[(ct*16+ln)*16], 0, lane), acc[ct]);
    __syncthreads();
    buf ^= 1;
  }
  if(R){
    #pragma unroll
    for(int ct=0;ct<4;++ct){
      const float* rp = R + (size_t)(rowBase + wv*16 + 8*half)*N + colBase + ct*16 + ln;
      #pragma unroll
      for(int r=0;r<8;++r) acc[ct][r] += rp[(size_t)r*N];
    }
  }
  #pragma unroll
  for(int ct=0;ct<4;++ct){
    float* cp = C + (size_t)(rowBase + wv*16 + 8*half)*N + colBase + ct*16 + ln;
    #pragma unroll
    for(int r=0;r<8;++r) cp[(size_t)r*N] = acc[ct][r];
  }
}

// ---------------- RoPE for Q: (B,S,NH,HD) f32 -> (B,NH,S,HD) bf16 ----------------
__global__ void k_rope_q(const float* __restrict__ q, const float* __restrict__ cosT,
                         const float* __restrict__ sinT, unsigned short* __restrict__ qbf){
  int idx = blockIdx.x*blockDim.x + threadIdx.x;       // B*S*NH*32
  if(idx >= BATCH*SEQ*NHEADS*32) return;
  int d = idx & 31;
  int h = (idx >> 5) & 15;
  int s = (idx >> 9) & 1023;
  int b = idx >> 19;
  const float* row = q + ((size_t)(b*SEQ + s))*HDIM + h*HEADD;
  float x1 = row[d], x2 = row[d+32];
  float o1 = x1*cosT[s*HEADD + d]      - x2*sinT[s*HEADD + d];
  float o2 = x2*cosT[s*HEADD + d + 32] + x1*sinT[s*HEADD + d + 32];
  size_t o = ((size_t)((b*NHEADS + h)*SEQ + s))*HEADD;
  qbf[o + d]      = f2bf(o1);
  qbf[o + d + 32] = f2bf(o2);
}

// ---------------- RoPE K + copy V: -> (B,NKV,S,HD) bf16 and fp32 present_kv ----------------
__global__ void k_rope_kv(const float* __restrict__ kf, const float* __restrict__ vf,
                          const float* __restrict__ cosT, const float* __restrict__ sinT,
                          unsigned short* __restrict__ kbf, unsigned short* __restrict__ vbf,
                          float* __restrict__ pk, float* __restrict__ pv){
  int idx = blockIdx.x*blockDim.x + threadIdx.x;       // B*S*NKV*32
  if(idx >= BATCH*SEQ*NKVH*32) return;
  int d = idx & 31;
  int h = (idx >> 5) & 3;
  int s = (idx >> 7) & 1023;
  int b = idx >> 17;
  const float* krow = kf + ((size_t)(b*SEQ + s))*(NKVH*HEADD) + h*HEADD;
  const float* vrow = vf + ((size_t)(b*SEQ + s))*(NKVH*HEADD) + h*HEADD;
  size_t o = ((size_t)((b*NKVH + h)*SEQ + s))*HEADD;
  float x1 = krow[d], x2 = krow[d+32];
  float o1 = x1*cosT[s*HEADD + d]      - x2*sinT[s*HEADD + d];
  float o2 = x2*cosT[s*HEADD + d + 32] + x1*sinT[s*HEADD + d + 32];
  kbf[o + d] = f2bf(o1); kbf[o + d + 32] = f2bf(o2);
  pk [o + d] = o1;       pk [o + d + 32] = o2;
  float v1 = vrow[d], v2 = vrow[d+32];
  vbf[o + d] = f2bf(v1); vbf[o + d + 32] = f2bf(v2);
  pv [o + d] = v1;       pv [o + d + 32] = v2;
}

// ---------------- flash attention: one block per (qtile64, head, batch) ----------------
__global__ __launch_bounds__(128)
void k_attn(const unsigned short* __restrict__ Qm, const unsigned short* __restrict__ Km,
            const unsigned* __restrict__ Vp, unsigned short* __restrict__ Om)
{
  __shared__ unsigned qa [64*32];
  __shared__ unsigned ka [64*32];
  __shared__ unsigned va [64*32];
  __shared__ unsigned pa_[64*32];
  __shared__ float sS[64*64];
  __shared__ float mrow[64], lrow[64], arw[64];

  const int qt = blockIdx.x, h = blockIdx.y, b = blockIdx.z;
  const int kvh = h >> 2;                              // GQA: 4 Q heads per KV head
  const int tid = threadIdx.x, lane = tid & 31, wv = tid >> 5;
  const int ln = lane & 15, half = lane >> 4;

  const unsigned* Qg = (const unsigned*)(Qm + (((size_t)(b*NHEADS + h)*SEQ) + qt*64)*HEADD);
  for(int i=tid;i<2048;i+=128) qa[i] = Qg[i];
  if(tid < 64){ mrow[tid] = -1e30f; lrow[tid] = 0.0f; }

  const unsigned* Vpb = Vp + (size_t)(b*NKVH + kvh)*(HEADD*(SEQ>>1));

  v8f oacc[4];
  #pragma unroll
  for(int i=0;i<4;++i) oacc[i] = zero8();

  for(int kt=0; kt<=qt; ++kt){
    const unsigned* Kg = (const unsigned*)(Km + (((size_t)(b*NKVH + kvh)*SEQ) + kt*64)*HEADD);
    for(int i=tid;i<2048;i+=128) ka[i] = Kg[i];
    for(int i=tid;i<2048;i+=128){
      int d = i >> 5, kp = i & 31;                     // pre-packed V (B-fragment layout)
      va[i] = Vpb[(size_t)d*(SEQ>>1) + kt*32 + kp];
    }
    __syncthreads();

    // S = Q @ K^T (scaled, causal-masked) -> LDS
    const unsigned* arowp = &qa[(wv*16 + ln)*32];
    v16bf a0 = frag_a(arowp, 0, lane);
    v16bf a1 = frag_a(arowp, 32, lane);
    #pragma unroll
    for(int ct=0;ct<4;++ct){
      const unsigned* bcol = &ka[(ct*16 + ln)*32];
      v8f s = zero8();
      s = wmma_bf16(a0, frag_b(bcol, 0,  lane), s);
      s = wmma_bf16(a1, frag_b(bcol, 32, lane), s);
      #pragma unroll
      for(int r=0;r<8;++r){
        int rt = wv*16 + r + 8*half;
        int grow = qt*64 + rt, gcol = kt*64 + ct*16 + ln;
        sS[rt*64 + ct*16 + ln] = s[r]*0.125f + (gcol <= grow ? 0.0f : -1e9f);
      }
    }
    __syncthreads();

    // online softmax per row (fp32), P -> bf16 A-layout
    if(tid < 64){
      int r = tid;
      float mo = mrow[r], rm = -1e30f;
      for(int j=0;j<64;++j) rm = fmaxf(rm, sS[r*64+j]);
      float mn = fmaxf(mo, rm);
      float al = __expf(mo - mn);
      float sum = 0.0f;
      for(int jp=0;jp<32;++jp){
        float p0 = __expf(sS[r*64 + 2*jp    ] - mn);
        float p1 = __expf(sS[r*64 + 2*jp + 1] - mn);
        sum += p0 + p1;
        pa_[r*32 + jp] = packbf(p0, p1);
      }
      lrow[r] = lrow[r]*al + sum;
      mrow[r] = mn;
      arw[r]  = al;
    }
    __syncthreads();

    // O = O*alpha + P @ V
    #pragma unroll
    for(int ct=0;ct<4;++ct){
      #pragma unroll
      for(int r=0;r<8;++r) oacc[ct][r] *= arw[wv*16 + r + 8*half];
    }
    const unsigned* prow = &pa_[(wv*16 + ln)*32];
    #pragma unroll
    for(int ks=0; ks<2; ++ks){
      v16bf ap = frag_a(prow, ks*32, lane);
      #pragma unroll
      for(int ct=0;ct<4;++ct)
        oacc[ct] = wmma_bf16(ap, frag_b(&va[(ct*16 + ln)*32], ks*32, lane), oacc[ct]);
    }
    __syncthreads();
  }

  #pragma unroll
  for(int ct=0;ct<4;++ct){
    #pragma unroll
    for(int r=0;r<8;++r){
      int rt = wv*16 + r + 8*half;
      float v = oacc[ct][r] / lrow[rt];
      size_t o = ((size_t)(b*SEQ) + qt*64 + rt)*HDIM + h*HEADD + ct*16 + ln;
      Om[o] = f2bf(v);
    }
  }
}

// ---------------- router logits (fp32 activations for selection fidelity) ----------------
__global__ void k_router(const float* __restrict__ h2, const float* __restrict__ rw,
                         float* __restrict__ logits){
  int idx = blockIdx.x*blockDim.x + threadIdx.x;      // NTOK*NEXP
  if(idx >= NTOK*NEXP) return;
  int e = idx & 7, t = idx >> 3;
  const float* xr = h2 + (size_t)t*HDIM;
  float acc = 0.0f;
  for(int i=0;i<HDIM;++i) acc += xr[i]*rw[i*NEXP + e];
  logits[idx] = acc;
}

// ---------------- softmax + top-2 + gather lists ----------------
__global__ void k_topk(const float* __restrict__ logits, float* __restrict__ probs,
                       int* __restrict__ cnt, int* __restrict__ gtok, float* __restrict__ gw2){
  int t = blockIdx.x*blockDim.x + threadIdx.x;
  if(t >= NTOK) return;
  float l[NEXP]; float m = -1e30f;
  for(int e=0;e<NEXP;++e){ l[e] = logits[t*NEXP + e]; m = fmaxf(m, l[e]); }
  float s = 0.0f;
  for(int e=0;e<NEXP;++e){ l[e] = __expf(l[e]-m); s += l[e]; }
  for(int e=0;e<NEXP;++e){ l[e] /= s; probs[t*NEXP + e] = l[e]; }
  int e0 = 0;
  for(int e=1;e<NEXP;++e) if(l[e] > l[e0]) e0 = e;
  int e1 = (e0==0) ? 1 : 0;
  for(int e=0;e<NEXP;++e) if(e!=e0 && l[e] > l[e1]) e1 = e;
  float z1 = __expf(l[e1]-l[e0]);                     // softmax over the two top probs
  float w0 = 1.0f/(1.0f+z1), w1 = z1/(1.0f+z1);
  int s0 = atomicAdd(&cnt[e0], 1); gtok[e0*NTOK+s0] = t; gw2[e0*NTOK+s0] = w0;
  int s1 = atomicAdd(&cnt[e1], 1); gtok[e1*NTOK+s1] = t; gw2[e1*NTOK+s1] = w1;
}

// ---------------- aux loss: E * sum_e cnt[e]/n * mean_t(probs[:,e]) ----------------
__global__ void k_aux(const float* __restrict__ probs, const int* __restrict__ cnt,
                      float* __restrict__ auxOut){
  __shared__ float red[256];
  float acc = 0.0f;
  for(int t=threadIdx.x; t<NTOK; t+=256)
    for(int e=0;e<NEXP;++e) acc += (float)cnt[e]*probs[t*NEXP + e];
  red[threadIdx.x] = acc; __syncthreads();
  for(int off=128; off>0; off>>=1){
    if((int)threadIdx.x < off) red[threadIdx.x] += red[threadIdx.x+off];
    __syncthreads();
  }
  if(threadIdx.x==0) auxOut[0] = red[0]*(float)NEXP/((float)NTOK*(float)NTOK);
}

// ---------------- fused gather gate/up + SwiGLU (per expert, packed weights) ----------------
__global__ __launch_bounds__(128)
void k_gateup(const unsigned short* __restrict__ X, const int* __restrict__ toks,
              const int* __restrict__ cntp, const unsigned* __restrict__ Wgp,
              const unsigned* __restrict__ Wup, unsigned short* __restrict__ Hout,
              int K, int N)
{
  const int count = *cntp;
  const int rowBase = blockIdx.y*64;
  if(rowBase >= count) return;
  __shared__ unsigned aL[1024];
  __shared__ unsigned gL[2][1024];
  __shared__ unsigned uL[2][1024];
  const int tid = threadIdx.x, lane = tid & 31, wv = tid >> 5;
  const int ln = lane & 15, half = lane >> 4;
  const int colBase = blockIdx.x*64;
  const int Kp = K >> 1;
  v8f g[4], u[4];
  #pragma unroll
  for(int i=0;i<4;++i){ g[i]=zero8(); u[i]=zero8(); }
  const int ktn = K >> 5;
  int buf = 0;
#if USE_TDM
  if(wv==0){
    tdm_load_tile64(lds_off(&gL[0][0]), Wgp + (size_t)colBase*Kp, (unsigned)Kp, (unsigned)N);
    tdm_load_tile64(lds_off(&uL[0][0]), Wup + (size_t)colBase*Kp, (unsigned)Kp, (unsigned)N);
  }
#endif
  for(int kt=0; kt<ktn; ++kt){
    for(int i=tid;i<1024;i+=128){
      int r = i>>4, c = i&15;
      int slot = rowBase + r;
      unsigned v = 0u;
      if(slot < count){
        int tok = toks[slot];
        v = *(const unsigned*)(X + (size_t)tok*K + kt*32 + 2*c);
      }
      aL[i] = v;
    }
#if USE_TDM
    if(wv==0) __builtin_amdgcn_s_wait_tensorcnt(0);
#else
    for(int i=tid;i<1024;i+=128){
      int n = i>>4, c = i&15;
      gL[0][i] = Wgp[(size_t)(colBase+n)*Kp + kt*16 + c];
      uL[0][i] = Wup[(size_t)(colBase+n)*Kp + kt*16 + c];
    }
#endif
    __syncthreads();
#if USE_TDM
    if(wv==0 && kt+1 < ktn){
      tdm_load_tile64(lds_off(&gL[buf^1][0]), Wgp + (size_t)colBase*Kp + (kt+1)*16, (unsigned)Kp, (unsigned)N);
      tdm_load_tile64(lds_off(&uL[buf^1][0]), Wup + (size_t)colBase*Kp + (kt+1)*16, (unsigned)Kp, (unsigned)N);
    }
    const unsigned* gb = gL[buf]; const unsigned* ub = uL[buf];
#else
    const unsigned* gb = gL[0];  const unsigned* ub = uL[0];
#endif
    v16bf a = frag_a(&aL[(wv*16+ln)*16], 0, lane);
    #pragma unroll
    for(int ct=0;ct<4;++ct){
      g[ct] = wmma_bf16(a, frag_b(&gb[(ct*16+ln)*16], 0, lane), g[ct]);
      u[ct] = wmma_bf16(a, frag_b(&ub[(ct*16+ln)*16], 0, lane), u[ct]);
    }
    __syncthreads();
    buf ^= 1;
  }
  #pragma unroll
  for(int ct=0;ct<4;++ct){
    int col = colBase + ct*16 + ln;
    #pragma unroll
    for(int r=0;r<8;++r){
      int srow = rowBase + wv*16 + r + 8*half;
      if(srow < count){
        float gg = g[ct][r], uu = u[ct][r];
        float hv = (gg/(1.0f + __expf(-gg)))*uu;       // silu(g)*u
        Hout[(size_t)srow*N + col] = f2bf(hv);
      }
    }
  }
}

// ---------------- down proj with weighted scatter-add into output ----------------
__global__ __launch_bounds__(128)
void k_down(const unsigned short* __restrict__ Hin, const int* __restrict__ toks,
            const int* __restrict__ cntp, const float* __restrict__ gwv,
            const unsigned* __restrict__ Wdp, float* __restrict__ Out,
            int K, int N)
{
  const int count = *cntp;
  const int rowBase = blockIdx.y*64;
  if(rowBase >= count) return;
  __shared__ unsigned aL[1024];
  __shared__ unsigned bL[2][1024];
  const int tid = threadIdx.x, lane = tid & 31, wv = tid >> 5;
  const int ln = lane & 15, half = lane >> 4;
  const int colBase = blockIdx.x*64;
  const int Kp = K >> 1;
  v8f acc[4];
  #pragma unroll
  for(int i=0;i<4;++i) acc[i] = zero8();
  const int ktn = K >> 5;
  int buf = 0;
#if USE_TDM
  if(wv==0)
    tdm_load_tile64(lds_off(&bL[0][0]), Wdp + (size_t)colBase*Kp, (unsigned)Kp, (unsigned)N);
#endif
  for(int kt=0; kt<ktn; ++kt){
    for(int i=tid;i<1024;i+=128){
      int r = i>>4, c = i&15;
      int slot = rowBase + r;
      unsigned v = 0u;
      if(slot < count) v = *(const unsigned*)(Hin + (size_t)slot*K + kt*32 + 2*c);
      aL[i] = v;
    }
#if USE_TDM
    if(wv==0) __builtin_amdgcn_s_wait_tensorcnt(0);
#else
    for(int i=tid;i<1024;i+=128){
      int n = i>>4, c = i&15;
      bL[0][i] = Wdp[(size_t)(colBase+n)*Kp + kt*16 + c];
    }
#endif
    __syncthreads();
#if USE_TDM
    if(wv==0 && kt+1 < ktn)
      tdm_load_tile64(lds_off(&bL[buf^1][0]), Wdp + (size_t)colBase*Kp + (kt+1)*16, (unsigned)Kp, (unsigned)N);
    const unsigned* bb = bL[buf];
#else
    const unsigned* bb = bL[0];
#endif
    v16bf a = frag_a(&aL[(wv*16+ln)*16], 0, lane);
    #pragma unroll
    for(int ct=0;ct<4;++ct)
      acc[ct] = wmma_bf16(a, frag_b(&bb[(ct*16+ln)*16], 0, lane), acc[ct]);
    __syncthreads();
    buf ^= 1;
  }
  #pragma unroll
  for(int ct=0;ct<4;++ct){
    int col = colBase + ct*16 + ln;
    #pragma unroll
    for(int r=0;r<8;++r){
      int srow = rowBase + wv*16 + r + 8*half;
      if(srow < count){
        int tok = toks[srow];
        Out[(size_t)tok*N + col] += gwv[srow]*acc[ct][r];   // serialized per-expert launches
      }
    }
  }
}

// ---------------- host orchestration ----------------
static inline size_t alignup(size_t x){ return (x + 255) & ~(size_t)255; }

extern "C" void kernel_launch(void* const* d_in, const int* in_sizes, int n_in,
                              void* d_out, int out_size, void* d_ws, size_t ws_size,
                              hipStream_t stream){
  (void)in_sizes; (void)n_in; (void)out_size; (void)ws_size;
  const float* x_in   = (const float*)d_in[0];
  // d_in[1] attn_mask (pure causal -1e9: applied analytically), d_in[2] position_ids (arange)
  const float* cosT   = (const float*)d_in[3];
  const float* sinT   = (const float*)d_in[4];
  const float* wq     = (const float*)d_in[5];
  const float* wk     = (const float*)d_in[6];
  const float* wv     = (const float*)d_in[7];
  const float* wo     = (const float*)d_in[8];
  const float* n1w    = (const float*)d_in[9];
  const float* n2w    = (const float*)d_in[10];
  const float* rout   = (const float*)d_in[11];
  const float* wg     = (const float*)d_in[12];
  const float* wu     = (const float*)d_in[13];
  const float* wd     = (const float*)d_in[14];

  char* base = (char*)d_ws; size_t off = 0;
  auto alloc = [&](size_t bytes)->void*{ void* r = base + off; off = alignup(off + bytes); return r; };

  unsigned*       wq_p  = (unsigned*)alloc((size_t)(HDIM/2)*HDIM*4);
  unsigned*       wk_p  = (unsigned*)alloc((size_t)(HDIM/2)*NKVH*HEADD*4);
  unsigned*       wv_p  = (unsigned*)alloc((size_t)(HDIM/2)*NKVH*HEADD*4);
  unsigned*       wo_p  = (unsigned*)alloc((size_t)(HDIM/2)*HDIM*4);
  unsigned*       wg_p  = (unsigned*)alloc((size_t)NEXP*(HDIM/2)*FDIM*4);
  unsigned*       wu_p  = (unsigned*)alloc((size_t)NEXP*(HDIM/2)*FDIM*4);
  unsigned*       wd_p  = (unsigned*)alloc((size_t)NEXP*(FDIM/2)*HDIM*4);
  unsigned short* h1_bf = (unsigned short*)alloc((size_t)NTOK*HDIM*2);
  float*          qbuf  = (float*)alloc((size_t)NTOK*HDIM*4);
  float*          kbuf  = (float*)alloc((size_t)NTOK*NKVH*HEADD*4);
  float*          vbuf  = (float*)alloc((size_t)NTOK*NKVH*HEADD*4);
  unsigned short* q_bf  = (unsigned short*)alloc((size_t)NTOK*HDIM*2);
  unsigned short* k_bf  = (unsigned short*)alloc((size_t)NTOK*NKVH*HEADD*2);
  unsigned short* v_bf  = (unsigned short*)alloc((size_t)NTOK*NKVH*HEADD*2);
  unsigned*       v_p   = (unsigned*)alloc((size_t)BATCH*NKVH*(SEQ/2)*HEADD*4);
  unsigned short* a_bf  = (unsigned short*)alloc((size_t)NTOK*HDIM*2);
  float*          x1    = (float*)alloc((size_t)NTOK*HDIM*4);
  unsigned short* h2_bf = (unsigned short*)alloc((size_t)NTOK*HDIM*2);
  float*          h2_f  = (float*)alloc((size_t)NTOK*HDIM*4);
  float*          logit = (float*)alloc((size_t)NTOK*NEXP*4);
  float*          probs = (float*)alloc((size_t)NTOK*NEXP*4);
  int*            cnt   = (int*)alloc(NEXP*sizeof(int));
  int*            gtok  = (int*)alloc((size_t)NEXP*NTOK*4);
  float*          gw2   = (float*)alloc((size_t)NEXP*NTOK*4);
  unsigned short* Hbuf  = (unsigned short*)alloc((size_t)NTOK*FDIM*2);

  auto packW = [&](const float* s, unsigned* d, int K, int N, int batch){
    int per = (K/2)*N;
    k_packT_f32<<<dim3((unsigned)((per + 255)/256), (unsigned)batch), dim3(256), 0, stream>>>(s, d, K, N);
  };
  packW(wq, wq_p, HDIM, HDIM, 1);
  packW(wk, wk_p, HDIM, NKVH*HEADD, 1);
  packW(wv, wv_p, HDIM, NKVH*HEADD, 1);
  packW(wo, wo_p, HDIM, HDIM, 1);
  packW(wg, wg_p, HDIM, FDIM, NEXP);
  packW(wu, wu_p, HDIM, FDIM, NEXP);
  packW(wd, wd_p, FDIM, HDIM, NEXP);

  float* out_x  = (float*)d_out;
  float* out_pk = out_x  + (size_t)BATCH*SEQ*HDIM;
  float* out_pv = out_pk + (size_t)BATCH*NKVH*SEQ*HEADD;
  float* out_ax = out_pv + (size_t)BATCH*NKVH*SEQ*HEADD;

  // ---- attention half ----
  k_rmsnorm<<<dim3(NTOK), dim3(256), 0, stream>>>(x_in, n1w, h1_bf, nullptr, HDIM);
  k_gemm<<<dim3(HDIM/64, NTOK/64), dim3(128), 0, stream>>>(h1_bf, wq_p, qbuf, nullptr, NTOK, HDIM, HDIM);
  k_gemm<<<dim3((NKVH*HEADD)/64, NTOK/64), dim3(128), 0, stream>>>(h1_bf, wk_p, kbuf, nullptr, NTOK, NKVH*HEADD, HDIM);
  k_gemm<<<dim3((NKVH*HEADD)/64, NTOK/64), dim3(128), 0, stream>>>(h1_bf, wv_p, vbuf, nullptr, NTOK, NKVH*HEADD, HDIM);
  k_rope_q<<<dim3((BATCH*SEQ*NHEADS*32)/256), dim3(256), 0, stream>>>(qbuf, cosT, sinT, q_bf);
  k_rope_kv<<<dim3((BATCH*SEQ*NKVH*32)/256), dim3(256), 0, stream>>>(kbuf, vbuf, cosT, sinT, k_bf, v_bf, out_pk, out_pv);
  {
    int per = (SEQ/2)*HEADD;
    k_packT_bf16<<<dim3((unsigned)((per + 255)/256), BATCH*NKVH), dim3(256), 0, stream>>>(v_bf, v_p, SEQ, HEADD);
  }
  k_attn<<<dim3(SEQ/64, NHEADS, BATCH), dim3(128), 0, stream>>>(q_bf, k_bf, v_p, a_bf);
  k_gemm<<<dim3(HDIM/64, NTOK/64), dim3(128), 0, stream>>>(a_bf, wo_p, x1, x_in, NTOK, HDIM, HDIM);
  (void)hipMemcpyAsync(out_x, x1, (size_t)NTOK*HDIM*4, hipMemcpyDeviceToDevice, stream);

  // ---- MoE half ----
  k_rmsnorm<<<dim3(NTOK), dim3(256), 0, stream>>>(x1, n2w, h2_bf, h2_f, HDIM);
  k_router<<<dim3((NTOK*NEXP)/256), dim3(256), 0, stream>>>(h2_f, rout, logit);
  (void)hipMemsetAsync(cnt, 0, NEXP*sizeof(int), stream);
  k_topk<<<dim3(NTOK/256), dim3(256), 0, stream>>>(logit, probs, cnt, gtok, gw2);
  k_aux<<<dim3(1), dim3(256), 0, stream>>>(probs, cnt, out_ax);

  for(int e=0;e<NEXP;++e){
    k_gateup<<<dim3(FDIM/64, NTOK/64), dim3(128), 0, stream>>>(
        h2_bf, gtok + (size_t)e*NTOK, cnt + e,
        wg_p + (size_t)e*(HDIM/2)*FDIM, wu_p + (size_t)e*(HDIM/2)*FDIM,
        Hbuf, HDIM, FDIM);
    k_down<<<dim3(HDIM/64, NTOK/64), dim3(128), 0, stream>>>(
        Hbuf, gtok + (size_t)e*NTOK, cnt + e, gw2 + (size_t)e*NTOK,
        wd_p + (size_t)e*(FDIM/2)*HDIM, out_x, FDIM, HDIM);
  }
}
